// BottleneckAttn_18665927868967
// MI455X (gfx1250) — matrix operand order repaired and checked by reading, not compiled, not run on hardware
//
#include <hip/hip_runtime.h>
#include <hip/hip_bf16.h>

// ---------------------------------------------------------------------------
// BottleneckAttn for MI455X (gfx1250, wave32, WMMA).
//  B=2, C=256, H=W=64, 4 heads, d_qk=d_v=64, SCALE=1/8.
//  All big GEMMs via v_wmma_f32_16x16x32_bf16 (bf16 in, f32 accumulate).
//  Flash-attention main loop: never materializes the 8x4096x4096 attn matrix.
//  Fragment loads are batched (all fragments of a burst live in distinct
//  registers) so each WMMA burst pays LDS/VMEM latency once, not per-tile.
// ---------------------------------------------------------------------------

typedef __bf16 bf16_t;
typedef __attribute__((ext_vector_type(16))) bf16_t bf16x16;
typedef __attribute__((ext_vector_type(8)))  bf16_t bf16x8;
typedef __attribute__((ext_vector_type(8)))  float  f32x8;
typedef __attribute__((ext_vector_type(4)))  unsigned int u32x4;

#define HW   4096
#define CC   256
#define SCALE 0.125f

#define WMMA_BF16(A, B, C) \
  __builtin_amdgcn_wmma_f32_16x16x32_bf16(false, (A), false, (B), (short)0, (C), false, false)

__device__ __forceinline__ f32x8 zero8() {
  f32x8 z;
#pragma unroll
  for (int i = 0; i < 8; ++i) z[i] = 0.0f;
  return z;
}

// Assemble a 16-element bf16 fragment from two contiguous 16-byte chunks.
__device__ __forceinline__ bf16x16 load_frag2(const bf16_t* p0, const bf16_t* p1) {
  union { bf16x16 f; u32x4 q[2]; } u;
  u.q[0] = *(const u32x4*)p0;
  u.q[1] = *(const u32x4*)p1;
  return u.f;
}

// ---------------------------------------------------------------------------
// Kernel 1: qkv = W(768x256) @ X(256x4096) per batch, written directly in the
// layouts the attention kernel wants:
//   Q [bh][n][d]  (bf16)   (transposed store from C-fragments, packed b128)
//   KT[bh][m][d]  (bf16)
//   V [bh][d][m]  (bf16)
// WG = 256 thr (8 waves), tile 128(o) x 128(n), K staged 32-wide through LDS.
// ---------------------------------------------------------------------------
__global__ __launch_bounds__(256) void k_qkv(const float* __restrict__ X,
                                             const float* __restrict__ Wt,
                                             bf16_t* __restrict__ Q,
                                             bf16_t* __restrict__ KT,
                                             bf16_t* __restrict__ V) {
  __shared__ bf16_t XT[128][40];  // [n][c] transposed slab (+8 pad vs banks)

  const int tid = threadIdx.x;
  const int wv = tid >> 5;      // wave 0..7  -> o strip
  const int ln = tid & 31;
  const int l2 = ln >> 4;       // lane half
  const int lm = ln & 15;
  const int ob = blockIdx.x * 128;
  const int nb = blockIdx.y * 128;
  const int b  = blockIdx.z;

  f32x8 acc[8];
#pragma unroll
  for (int t = 0; t < 8; ++t) acc[t] = zero8();

  for (int ks = 0; ks < 8; ++ks) {
    const int cs = ks * 32;
    __syncthreads();
    // stage X[cs..cs+31][nb..nb+127] -> XT[n][c] (fp32 -> bf16, transpose)
#pragma unroll
    for (int p = 0; p < 4; ++p) {
      const int c_loc = p * 8 + (tid >> 5);
      const int n4 = (tid & 31) * 4;
      const float4 xv = *(const float4*)&X[(size_t)(b * CC + cs + c_loc) * HW + nb + n4];
      XT[n4 + 0][c_loc] = (bf16_t)xv.x;
      XT[n4 + 1][c_loc] = (bf16_t)xv.y;
      XT[n4 + 2][c_loc] = (bf16_t)xv.z;
      XT[n4 + 3][c_loc] = (bf16_t)xv.w;
    }
    __syncthreads();

    // A fragment: W rows (o), contiguous in c, straight from global (L2 hot)
    const int orow = ob + wv * 16 + lm;
    const float* wp = &Wt[(size_t)orow * CC + cs];
    bf16x16 af;
#pragma unroll
    for (int j = 0; j < 8; ++j) af[j] = (bf16_t)wp[8 * l2 + j];
#pragma unroll
    for (int j = 0; j < 8; ++j) af[8 + j] = (bf16_t)wp[16 + 8 * l2 + j];

    // Batch-load ALL 8 B fragments (distinct regs, can't be coalesced),
    // then an uninterrupted WMMA burst: one LDS wait per slab, not per tile.
    bf16x16 bfr[8];
#pragma unroll
    for (int t = 0; t < 8; ++t) {
      const bf16_t* bp = &XT[t * 16 + lm][8 * l2];
      bfr[t] = load_frag2(bp, bp + 16);
    }
#pragma unroll
    for (int t = 0; t < 8; ++t)
      acc[t] = WMMA_BF16(af, bfr[t], acc[t]);
  }

  // Epilogue: C-frag element v -> o = ob + 16*wv + 8*l2 + v ; col -> n
  const int obase8 = ob + wv * 16 + l2 * 8;
#pragma unroll
  for (int t = 0; t < 8; ++t) {
    const int n = nb + t * 16 + lm;
    if (ob < 512) {  // q or k -> (n, d) layout, 8 consecutive d per lane
      const int olocal = (ob < 256) ? obase8 : (obase8 - 256);
      const int head = olocal >> 6;
      const int d0 = olocal & 63;
      const int bh = b * 4 + head;
      bf16_t* dst = (ob < 256 ? Q : KT) + (((size_t)(bh * HW) + n) * 64 + d0);
      union { bf16x8 h; u32x4 q; } u;
#pragma unroll
      for (int v = 0; v < 8; ++v) u.h[v] = (bf16_t)acc[t][v];
      *(u32x4*)dst = u.q;
    } else {  // v -> (d, m) layout
#pragma unroll
      for (int v = 0; v < 8; ++v) {
        const int o = obase8 - 512 + v;
        const int head = o >> 6, d = o & 63;
        const int bh = b * 4 + head;
        V[((size_t)(bh * 64 + d)) * HW + n] = (bf16_t)acc[t][v];
      }
    }
  }
}

// ---------------------------------------------------------------------------
// Kernel 2: RH[bh][n][r] = Q[bh,n,:]·height_rel[r,:]  (r<127, pitch 128)
//           RW[bh][n][r] = Q[bh,n,:]·width_rel[r,:]
// rel_k is (127,64) fp32 = exactly (N,K) row-major for B-fragments.
// ---------------------------------------------------------------------------
__global__ __launch_bounds__(128) void k_rel(const bf16_t* __restrict__ Q,
                                             const float* __restrict__ hrel,
                                             const float* __restrict__ wrel,
                                             float* __restrict__ RH,
                                             float* __restrict__ RW) {
  const int tid = threadIdx.x;
  const int wv = tid >> 5, ln = tid & 31, l2 = ln >> 4, lm = ln & 15;
  const int bh = blockIdx.y;
  const int n0 = blockIdx.x * 64 + wv * 16;

  bf16x16 qa[2];
#pragma unroll
  for (int ks = 0; ks < 2; ++ks) {
    const bf16_t* qp = &Q[((size_t)(bh * HW) + n0 + lm) * 64 + ks * 32];
    qa[ks] = load_frag2(qp + 16 * l2, qp + 16 * l2 + 8);
  }

#pragma unroll
  for (int which = 0; which < 2; ++which) {
    const float* rel = which ? wrel : hrel;
    float* out = which ? RW : RH;
#pragma unroll
    for (int t = 0; t < 8; ++t) {
      int r = t * 16 + lm;
      if (r > 126) r = 126;  // clamp; column 127 is written but never read
      f32x8 acc = zero8();
#pragma unroll
      for (int ks = 0; ks < 2; ++ks) {
        const float* rp = &rel[(size_t)r * 64 + ks * 32 + 16 * l2];
        bf16x16 bfr;
#pragma unroll
        for (int j = 0; j < 16; ++j) bfr[j] = (bf16_t)rp[j];
        acc = WMMA_BF16(qa[ks], bfr, acc);
      }
#pragma unroll
      for (int v = 0; v < 8; ++v) {
        const int n = n0 + l2 * 8 + v;
        out[((size_t)(bh * HW) + n) * 128 + t * 16 + lm] = acc[v];
      }
    }
  }
}

// ---------------------------------------------------------------------------
// Kernel 3: flash attention.
//  Grid (64 n-tiles, 8 bh), WG = 128 thr (4 waves), 64 query rows per WG
//  (aligned to W=64 so h is constant and w == local row).
//  Per 64-key block (aligned -> h2 == iter):
//    - batch-load all 8 Kt B-frags, S = Q Kt as one 8-WMMA burst,
//    - V B-frags loaded next so their global latency overlaps softmax VALU,
//    - bias = RHl[it][row] (LDS) + rwreg (regs), online softmax in place,
//    - P -> LDS (bf16, wave-private rows), O += P V as one 8-WMMA burst.
// ---------------------------------------------------------------------------
__global__ __launch_bounds__(128) void k_attn(const bf16_t* __restrict__ Q,
                                              const bf16_t* __restrict__ KT,
                                              const bf16_t* __restrict__ V,
                                              const float* __restrict__ RH,
                                              const float* __restrict__ RW,
                                              float* __restrict__ Out) {
  __shared__ float  RHl[64][64];  // [it][row] : RH[bh, nb+row, it+63-ny]
  __shared__ bf16_t Pl[64][72];   // softmax probs staging (pitch pad 8)

  const int tid = threadIdx.x;
  const int wv = tid >> 5, ln = tid & 31, l2 = ln >> 4, lm = ln & 15;
  const int ny = blockIdx.x;      // 0..63 : h of this query tile
  const int bh = blockIdx.y;      // 0..7
  const int nb = ny * 64;

  // Shifted h-bias table: one value per (row, key-block)
  for (int i = tid; i < 64 * 64; i += 128) {
    const int it = i & 63, row = i >> 6;
    RHl[it][row] = RH[((size_t)(bh * HW) + nb + row) * 128 + it + 63 - ny];
  }
  __syncthreads();

  // w-bias is iteration-invariant per lane element: hoist into registers
  float rwreg[4][8];
#pragma unroll
  for (int c = 0; c < 4; ++c) {
    const int w2 = c * 16 + lm;
#pragma unroll
    for (int v = 0; v < 8; ++v) {
      const int row = wv * 16 + l2 * 8 + v;  // == w of this query row
      rwreg[c][v] = RW[((size_t)(bh * HW) + nb + row) * 128 + (w2 + 63 - row)];
    }
  }

  // Q A-fragments, loaded once, live in VGPRs for all 64 key blocks
  bf16x16 qa[2];
#pragma unroll
  for (int ks = 0; ks < 2; ++ks) {
    const bf16_t* qp = &Q[((size_t)(bh * HW) + nb + wv * 16 + lm) * 64 + ks * 32];
    qa[ks] = load_frag2(qp + 16 * l2, qp + 16 * l2 + 8);
  }

  float mrun[8], lsum[8];
  f32x8 o[4];
#pragma unroll
  for (int v = 0; v < 8; ++v) { mrun[v] = -3.0e38f; lsum[v] = 0.0f; }
#pragma unroll
  for (int dt = 0; dt < 4; ++dt) o[dt] = zero8();

  for (int it = 0; it < 64; ++it) {
    const int m0 = it * 64;

    if (it + 1 < 64) {  // pull next key block toward L0/L2
      __builtin_prefetch(&KT[((size_t)(bh * HW) + m0 + 64 + ln) * 64], 0, 1);
      __builtin_prefetch(&V[((size_t)(bh * 64 + ln)) * HW + m0 + 64], 0, 1);
    }

    float rh[8];
    const float* rhp = &RHl[it][wv * 16 + l2 * 8];
#pragma unroll
    for (int v = 0; v < 8; ++v) rh[v] = rhp[v];

    // ---- S = Q Kt : batch-load all 8 Kt B-frags, then one WMMA burst ----
    const bf16_t* kbase = &KT[((size_t)(bh * HW) + m0 + lm) * 64 + 16 * l2];
    bf16x16 kb[8];
#pragma unroll
    for (int c = 0; c < 4; ++c) {
      const bf16_t* kp = kbase + (size_t)c * 16 * 64;
      kb[2 * c]     = *(const bf16x16*)(kp);
      kb[2 * c + 1] = *(const bf16x16*)(kp + 32);
    }
    f32x8 s[4];
#pragma unroll
    for (int c = 0; c < 4; ++c) {
      s[c] = zero8();
      s[c] = WMMA_BF16(qa[0], kb[2 * c], s[c]);
      s[c] = WMMA_BF16(qa[1], kb[2 * c + 1], s[c]);
    }

    // ---- hoist V B-fragments: global loads overlap the softmax phase ----
    bf16x16 vb[2][4];
#pragma unroll
    for (int ks = 0; ks < 2; ++ks)
#pragma unroll
      for (int dt = 0; dt < 4; ++dt) {
        const int d = dt * 16 + lm;
        vb[ks][dt] = *(const bf16x16*)&V[((size_t)(bh * 64 + d)) * HW + m0 + ks * 32 + 16 * l2];
      }

    // ---- logits + bias (in place) ----
#pragma unroll
    for (int c = 0; c < 4; ++c)
#pragma unroll
      for (int v = 0; v < 8; ++v)
        s[c][v] = s[c][v] * SCALE + rh[v] + rwreg[c][v];

    // ---- online softmax (in place: s becomes P) ----
    float alpha[8];
#pragma unroll
    for (int v = 0; v < 8; ++v) {
      float mv = fmaxf(fmaxf(s[0][v], s[1][v]), fmaxf(s[2][v], s[3][v]));
      mv = fmaxf(mv, __shfl_xor(mv, 1));
      mv = fmaxf(mv, __shfl_xor(mv, 2));
      mv = fmaxf(mv, __shfl_xor(mv, 4));
      mv = fmaxf(mv, __shfl_xor(mv, 8));
      const float mnew = fmaxf(mrun[v], mv);
      alpha[v] = __expf(mrun[v] - mnew);
      float srow = 0.0f;
#pragma unroll
      for (int c = 0; c < 4; ++c) {
        const float p = __expf(s[c][v] - mnew);
        s[c][v] = p;
        srow += p;
      }
      srow += __shfl_xor(srow, 1);
      srow += __shfl_xor(srow, 2);
      srow += __shfl_xor(srow, 4);
      srow += __shfl_xor(srow, 8);
      lsum[v] = lsum[v] * alpha[v] + srow;
      mrun[v] = mnew;
    }
#pragma unroll
    for (int dt = 0; dt < 4; ++dt)
#pragma unroll
      for (int v = 0; v < 8; ++v) o[dt][v] *= alpha[v];

    // ---- P -> LDS (each wave's rows are private to it; no barrier) ----
#pragma unroll
    for (int c = 0; c < 4; ++c)
#pragma unroll
      for (int v = 0; v < 8; ++v)
        Pl[wv * 16 + l2 * 8 + v][c * 16 + lm] = (bf16_t)s[c][v];

    // ---- O += P V : both P A-frags loaded, then one 8-WMMA burst ----
    const bf16_t* pp = &Pl[wv * 16 + lm][0];
    bf16x16 pa0 = load_frag2(pp + 16 * l2, pp + 16 * l2 + 8);
    bf16x16 pa1 = load_frag2(pp + 32 + 16 * l2, pp + 40 + 16 * l2);
#pragma unroll
    for (int dt = 0; dt < 4; ++dt)
      o[dt] = WMMA_BF16(pa0, vb[0][dt], o[dt]);
#pragma unroll
    for (int dt = 0; dt < 4; ++dt)
      o[dt] = WMMA_BF16(pa1, vb[1][dt], o[dt]);
  }

  // ---- epilogue: out[b, head*64+d, n] = O / l ----
  const int head = bh & 3, b = bh >> 2;
  float inv[8];
#pragma unroll
  for (int v = 0; v < 8; ++v) inv[v] = 1.0f / lsum[v];
#pragma unroll
  for (int dt = 0; dt < 4; ++dt) {
    const int d = dt * 16 + lm;
    const int n0o = nb + wv * 16 + l2 * 8;
    float* op = &Out[((size_t)(b * 256 + head * 64 + d)) * HW + n0o];
    *(float4*)(op + 0) = make_float4(o[dt][0] * inv[0], o[dt][1] * inv[1],
                                     o[dt][2] * inv[2], o[dt][3] * inv[3]);
    *(float4*)(op + 4) = make_float4(o[dt][4] * inv[4], o[dt][5] * inv[5],
                                     o[dt][6] * inv[6], o[dt][7] * inv[7]);
  }
}

// ---------------------------------------------------------------------------
extern "C" void kernel_launch(void* const* d_in, const int* in_sizes, int n_in,
                              void* d_out, int out_size, void* d_ws, size_t ws_size,
                              hipStream_t stream) {
  const float* x     = (const float*)d_in[0];  // (2,256,64,64)
  const float* qkv_w = (const float*)d_in[1];  // (768,256)
  const float* hrel  = (const float*)d_in[2];  // (127,64)  (already * SCALE)
  const float* wrel  = (const float*)d_in[3];  // (127,64)
  float* out = (float*)d_out;                  // (2,256,64,64)

  char* ws = (char*)d_ws;
  bf16_t* Q  = (bf16_t*)(ws + (size_t)0);           // 4 MB
  bf16_t* KT = (bf16_t*)(ws + ((size_t)4 << 20));   // 4 MB
  bf16_t* V  = (bf16_t*)(ws + ((size_t)8 << 20));   // 4 MB
  float*  RH = (float*)(ws + ((size_t)12 << 20));   // 16 MB
  float*  RW = (float*)(ws + ((size_t)28 << 20));   // 16 MB  (total 44 MB)

  k_qkv<<<dim3(6, 32, 2), 256, 0, stream>>>(x, qkv_w, Q, KT, V);
  k_rel<<<dim3(64, 8), 128, 0, stream>>>(Q, hrel, wrel, RH, RW);
  k_attn<<<dim3(64, 8), 128, 0, stream>>>(Q, KT, V, RH, RW, out);
}
